// DCM_22058952032783
// MI455X (gfx1250) — matrix operand order repaired and checked by reading, not compiled, not run on hardware
//
#include <hip/hip_runtime.h>

typedef __bf16 bf16;
typedef __attribute__((ext_vector_type(16))) __bf16 v16bf;
typedef __attribute__((ext_vector_type(8)))  float  v8f;
typedef __attribute__((ext_vector_type(4)))  float  v4f;
typedef __attribute__((ext_vector_type(4)))  __bf16 v4bf;
typedef __attribute__((ext_vector_type(4)))  unsigned int v4u;  // trivial 16B vector

union Frag { v16bf v; v4u q[2]; };

// ---------------------------------------------------------------------------
// fp32 -> bf16 conversion (4 elements/thread; all our sizes are multiples of 4)
// ---------------------------------------------------------------------------
__global__ void f32_to_bf16_kernel(const float* __restrict__ in, bf16* __restrict__ out, long n) {
  long i4 = ((long)blockIdx.x * blockDim.x + threadIdx.x) * 4;
  if (i4 >= n) return;
  v4f f = *(const v4f*)(in + i4);
  v4bf o;
  o.x = (bf16)f.x; o.y = (bf16)f.y; o.z = (bf16)f.z; o.w = (bf16)f.w;
  *(v4bf*)(out + i4) = o;
}

// ---------------------------------------------------------------------------
// adaptive average pool: y [8,512,60,60] -> out [8,512,k,k]; one 64-thread
// block per output element (k=1 averages 3600 values -> needs the parallelism)
// ---------------------------------------------------------------------------
__global__ void pool_kernel(const float* __restrict__ y, float* __restrict__ out, int k) {
  int ob = blockIdx.x;             // [0, 8*512*k*k)
  int kk = k * k;
  int ij = ob % kk, bc = ob / kk;
  int i = ij / k, j = ij % k;
  int s = 60 / k;
  const float* base = y + (long)bc * 3600 + (long)i * s * 60 + (long)j * s;
  int tot = s * s;
  float sum = 0.f;
  for (int t = threadIdx.x; t < tot; t += 64) {
    int r = t / s, c = t - r * s;
    sum += base[r * 60 + c];
  }
  __shared__ float red[64];
  red[threadIdx.x] = sum;
  __syncthreads();
#pragma unroll
  for (int off = 32; off > 0; off >>= 1) {
    if (threadIdx.x < off) red[threadIdx.x] += red[threadIdx.x + off];
    __syncthreads();
  }
  if (threadIdx.x == 0) out[ob] = red[0] / (float)tot;
}

// ---------------------------------------------------------------------------
// dynamic kernel generation: kern[b,m,ij] = gk_b[m] + sum_c gk_w[m,c]*pooled[b,c,ij]
// one 64-thread block per output element (512-long reduction)
// ---------------------------------------------------------------------------
__global__ void genkern_kernel(const float* __restrict__ gw, const float* __restrict__ gb,
                               const float* __restrict__ pooled, float* __restrict__ kern, int kk) {
  int ob = blockIdx.x;             // [0, 8*512*kk)
  int ij = ob % kk;
  int t2 = ob / kk;
  int m = t2 % 512, b = t2 / 512;
  const float* p = pooled + (long)b * 512 * kk + ij;
  const float* w = gw + (long)m * 512;
  float sum = 0.f;
  for (int c = threadIdx.x; c < 512; c += 64) sum += w[c] * p[(long)c * kk];
  __shared__ float red[64];
  red[threadIdx.x] = sum;
  __syncthreads();
#pragma unroll
  for (int off = 32; off > 0; off >>= 1) {
    if (threadIdx.x < off) red[threadIdx.x] += red[threadIdx.x + off];
    __syncthreads();
  }
  if (threadIdx.x == 0) kern[ob] = red[0] + gb[m];
}

// ---------------------------------------------------------------------------
// per-sample depthwise conv, k in {1,3,5}, pad k/2; in/out [8*512, 64, 64] bf16
// ---------------------------------------------------------------------------
__global__ void dwconv_kernel(const bf16* __restrict__ in, const float* __restrict__ kern,
                              bf16* __restrict__ out, int k) {
  int idx = blockIdx.x * blockDim.x + threadIdx.x;  // exactly 8*512*4096 threads
  int p = idx & 4095;
  int bm = idx >> 12;
  int yy = p >> 6, xx = p & 63;
  int pad = k >> 1;
  const bf16* ip = in + ((long)bm << 12);
  const float* kp = kern + (long)bm * k * k;
  float sum = 0.f;
  for (int dy = 0; dy < k; ++dy) {
    int sy = yy + dy - pad;
    if ((unsigned)sy > 63u) continue;
    for (int dx = 0; dx < k; ++dx) {
      int sx = xx + dx - pad;
      if ((unsigned)sx > 63u) continue;
      sum += kp[dy * k + dx] * (float)ip[sy * 64 + sx];
    }
  }
  out[idx] = (bf16)sum;
}

// ---------------------------------------------------------------------------
// Tiled bf16 WMMA GEMM: C[M,N] = A[M,K]*B[K,N] (+bias), batched over blockIdx.z.
// Block tile 128x128, 8 waves (2x4), wave tile 64x32, K-step 32.
// Double-buffered LDS; A tile staged with global_load_async_to_lds_b128
// (ASYNCcnt), B tile loaded k-major into registers and stored b128 into a
// padded [n][40] LDS layout (bank-conflict free for both stores and fragment
// reads: 80B row stride -> 20-dword lane stride, 16 disjoint 4-bank groups).
// ---------------------------------------------------------------------------
#define GSTRIDE 40  // padded LDS row stride in bf16 elements (80 bytes)

template <bool OUT_BF16, bool ACCUM, bool HAS_BIAS>
__global__ void __launch_bounds__(256) gemm_wmma_bf16(
    const bf16* __restrict__ A, int lda,
    const bf16* __restrict__ B, int ldb, long strideB,
    void* __restrict__ OutP, long strideOut, int ldout,
    const float* __restrict__ bias, int K) {
  // smem[0][buf] = A tile [m][k] (padded), smem[1][buf] = B tile [n][k] (padded)
  __shared__ alignas(16) bf16 smem[2][2][128 * GSTRIDE];

  const int tid  = threadIdx.x;
  const int wave = tid >> 5;
  const int lane = tid & 31;
  const int wm = wave >> 2;        // 0..1 -> M offset wm*64
  const int wn = wave & 3;         // 0..3 -> N offset wn*32
  const int hl = lane >> 4;        // half-wave
  const int l15 = lane & 15;
  const int tileM = blockIdx.y * 128;
  const int tileN = blockIdx.x * 128;
  const bf16* Bb = B + (long)blockIdx.z * strideB;

  // per-thread staging coordinates
  const int ar0 = tid >> 2, ac0 = (tid & 3) << 3;            // A chunk 0
  const int ar1 = (tid + 256) >> 2, ac1 = ((tid + 256) & 3) << 3;
  const int bn0 = tid & 127, bk0 = (tid >> 7) << 3;          // B chunk 0
  const int bn1 = bn0, bk1 = bk0 + 16;                       // B chunk 1 (c=tid+256)

  unsigned short breg[2][8];

  // ---- staging helpers -----------------------------------------------------
  auto stageA = [&](int kstep, int buf) {
    const bf16* g0 = A + (long)(tileM + ar0) * lda + kstep + ac0;
    const bf16* g1 = A + (long)(tileM + ar1) * lda + kstep + ac1;
    unsigned l0 = (unsigned)(size_t)&smem[0][buf][ar0 * GSTRIDE + ac0];
    unsigned l1 = (unsigned)(size_t)&smem[0][buf][ar1 * GSTRIDE + ac1];
    asm volatile("global_load_async_to_lds_b128 %0, %1, off" :: "v"(l0), "v"(g0) : "memory");
    asm volatile("global_load_async_to_lds_b128 %0, %1, off" :: "v"(l1), "v"(g1) : "memory");
  };
  auto loadB = [&](int kstep) {  // 8 coalesced u16 loads along k per chunk
#pragma unroll
    for (int it = 0; it < 2; ++it) {
      int kbase = kstep + (it ? bk1 : bk0);
      const unsigned short* g =
          (const unsigned short*)(Bb + (long)kbase * ldb + tileN + bn0);
#pragma unroll
      for (int j = 0; j < 8; ++j) breg[it][j] = g[(long)j * ldb];
    }
  };
  auto storeB = [&](int buf) {
#pragma unroll
    for (int it = 0; it < 2; ++it) {
      int kb = it ? bk1 : bk0;
      v4u pk;
      pk.x = (unsigned)breg[it][0] | ((unsigned)breg[it][1] << 16);
      pk.y = (unsigned)breg[it][2] | ((unsigned)breg[it][3] << 16);
      pk.z = (unsigned)breg[it][4] | ((unsigned)breg[it][5] << 16);
      pk.w = (unsigned)breg[it][6] | ((unsigned)breg[it][7] << 16);
      *(v4u*)&smem[1][buf][bn0 * GSTRIDE + kb] = pk;
    }
  };

  v8f acc[4][2] = {};

  // ---- pipelined main loop -------------------------------------------------
  const int nsteps = K >> 5;
  loadB(0);
  stageA(0, 0);
  storeB(0);
  asm volatile("s_wait_asynccnt 0x0" ::: "memory");
  __syncthreads();

  for (int s = 0; s < nsteps; ++s) {
    const int cur = s & 1;
    if (s + 1 < nsteps) {
      loadB((s + 1) << 5);        // global B in flight during WMMA phase
      stageA((s + 1) << 5, cur ^ 1);
    }

    Frag fa[4], fb[2];
#pragma unroll
    for (int mt = 0; mt < 4; ++mt) {
      int m = wm * 64 + mt * 16 + l15;
      fa[mt].q[0] = *(const v4u*)&smem[0][cur][m * GSTRIDE + 8 * hl];       // k = 8*hl..+7
      fa[mt].q[1] = *(const v4u*)&smem[0][cur][m * GSTRIDE + 16 + 8 * hl];  // k = 16+8*hl..+7
    }
#pragma unroll
    for (int nt = 0; nt < 2; ++nt) {
      int n = wn * 32 + nt * 16 + l15;
      fb[nt].q[0] = *(const v4u*)&smem[1][cur][n * GSTRIDE + 8 * hl];       // k = 8*hl..+7
      fb[nt].q[1] = *(const v4u*)&smem[1][cur][n * GSTRIDE + 16 + 8 * hl];  // k = 16+8*hl..+7
    }
#pragma unroll
    for (int mt = 0; mt < 4; ++mt)
#pragma unroll
      for (int nt = 0; nt < 2; ++nt)
        acc[mt][nt] = __builtin_amdgcn_wmma_f32_16x16x32_bf16(
            false, fa[mt].v, false, fb[nt].v, (short)0, acc[mt][nt], false, false);

    if (s + 1 < nsteps) storeB(cur ^ 1);  // waits for B loads only now
    asm volatile("s_wait_asynccnt 0x0" ::: "memory");
    __syncthreads();
  }

  // ---- epilogue: C VGPR v -> row v+8*hl, col l15 ---------------------------
#pragma unroll
  for (int mt = 0; mt < 4; ++mt) {
#pragma unroll
    for (int nt = 0; nt < 2; ++nt) {
      int col = tileN + wn * 32 + nt * 16 + l15;
#pragma unroll
      for (int v = 0; v < 8; ++v) {
        int row = tileM + wm * 64 + mt * 16 + v + 8 * hl;
        float val = acc[mt][nt][v];
        if (HAS_BIAS) val += bias[row];
        long off = (long)blockIdx.z * strideOut + (long)row * ldout + col;
        if (OUT_BF16) {
          ((bf16*)OutP)[off] = (bf16)val;
        } else {
          float* o = (float*)OutP;
          if (ACCUM) o[off] += val;
          else       o[off] = val;
        }
      }
    }
  }
}

// ---------------------------------------------------------------------------
// host-side orchestration
// ---------------------------------------------------------------------------
extern "C" void kernel_launch(void* const* d_in, const int* in_sizes, int n_in,
                              void* d_out, int out_size, void* d_ws, size_t ws_size,
                              hipStream_t stream) {
  (void)in_sizes; (void)n_in; (void)out_size; (void)ws_size;
  const float* x   = (const float*)d_in[0];
  const float* y   = (const float*)d_in[1];
  const float* gkw = (const float*)d_in[2];
  const float* gkb = (const float*)d_in[3];
  const float* trw = (const float*)d_in[4];
  const float* trb = (const float*)d_in[5];
  const float* fiw = (const float*)d_in[6];
  const float* fib = (const float*)d_in[7];
  const float* fow = (const float*)d_in[8];
  const float* fob = (const float*)d_in[9];
  float* out = (float*)d_out;

  char* ws = (char*)d_ws;
  size_t off = 0;
  auto carve = [&](size_t bytes) -> char* {
    char* p = ws + off;
    off = (off + bytes + 255) & ~(size_t)255;
    return p;
  };

  const long ACT = 8L * 512 * 4096;         // activation element count
  const long ACT_STRIDE = 512L * 4096;      // per-batch stride
  bf16* xb   = (bf16*)carve(ACT * 2);       // bf16(x)
  bf16* xin  = (bf16*)carve(ACT * 2);       // tr output (per branch, reused)
  bf16* dwb  = (bf16*)carve(ACT * 2);       // depthwise output (reused)
  bf16* fio  = (bf16*)carve(ACT * 2);       // fi output (reused)
  bf16* trwb = (bf16*)carve(3L * 512 * 512 * 2);
  bf16* fiwb = (bf16*)carve(3L * 512 * 512 * 2);
  bf16* fowb = (bf16*)carve(512L * 2048 * 2);
  const int KS[3] = {1, 3, 5};
  float* pooled[3];
  float* kernb[3];
  for (int i = 0; i < 3; ++i) {
    pooled[i] = (float*)carve(8L * 512 * KS[i] * KS[i] * 4);
    kernb[i]  = (float*)carve(8L * 512 * KS[i] * KS[i] * 4);
  }

  auto cvt = [&](const float* src, bf16* dst, long n) {
    f32_to_bf16_kernel<<<(unsigned)((n / 4 + 255) / 256), 256, 0, stream>>>(src, dst, n);
  };
  cvt(x,   xb,   ACT);
  cvt(trw, trwb, 3L * 512 * 512);
  cvt(fiw, fiwb, 3L * 512 * 512);
  cvt(fow, fowb, 512L * 2048);

  for (int br = 0; br < 3; ++br) {
    int k = KS[br], kk = k * k;
    int tot = 8 * 512 * kk;                 // one 64-thread block per output
    pool_kernel<<<tot, 64, 0, stream>>>(y, pooled[br], k);
    genkern_kernel<<<tot, 64, 0, stream>>>(
        gkw + (long)br * 512 * 512, gkb + br * 512, pooled[br], kernb[br], kk);
  }

  dim3 ggrid(4096 / 128, 512 / 128, 8);

  // out = fo_b + fo_w[:, 0:512] @ x      (fp32 store, with bias)
  gemm_wmma_bf16<false, false, true><<<ggrid, 256, 0, stream>>>(
      fowb, 2048, xb, 4096, ACT_STRIDE, out, ACT_STRIDE, 4096, fob, 512);

  for (int br = 0; br < 3; ++br) {
    int k = KS[br];
    // x_in = tr_w[br] @ x + tr_b[br]     (bf16 store)
    gemm_wmma_bf16<true, false, true><<<ggrid, 256, 0, stream>>>(
        trwb + (long)br * 512 * 512, 512, xb, 4096, ACT_STRIDE,
        xin, ACT_STRIDE, 4096, trb + br * 512, 512);
    // per-sample depthwise dynamic conv
    dwconv_kernel<<<(int)(ACT / 256), 256, 0, stream>>>(xin, kernb[br], dwb, k);
    // fi = fi_w[br] @ dw + fi_b[br]      (bf16 store)
    gemm_wmma_bf16<true, false, true><<<ggrid, 256, 0, stream>>>(
        fiwb + (long)br * 512 * 512, 512, dwb, 4096, ACT_STRIDE,
        fio, ACT_STRIDE, 4096, fib + br * 512, 512);
    // out += fo_w[:, 512*(br+1):512*(br+2)] @ fi   (fp32 accumulate)
    gemm_wmma_bf16<false, true, false><<<ggrid, 256, 0, stream>>>(
        fowb + 512L * (br + 1), 2048, fio, 4096, ACT_STRIDE,
        out, ACT_STRIDE, 4096, nullptr, 512);
  }
}
